// HDNNP_31035433681587
// MI455X (gfx1250) — matrix operand order repaired and checked by reading, not compiled, not run on hardware
//
#include <hip/hip_runtime.h>

// ---------------------------------------------------------------------------
// HDNNP forward for MI455X (gfx1250, wave32, WMMA, LDS-staged activations).
//
// Param flattening assumption (top-level dict insertion order, params pytree
// with dicts sorted by key):
//   0 pos, 1 edge_index, 2 atomic_numbers, 3 batch, 4 energy_shift,
//   5..22  ann['1'],['6'],['7'] each (W0,b0,W1,b1,W2,b2),
//   23     coeffs,
//   24..83 convs[0..4] each (lin_atom W,b, lin_nbr W,b, lin_out W,b,
//                            lin_res_1 W,b, lin_res_2 W,b, lin_rbf W,b),
//   84     embedding, 85..90 nn_ef (W0,b0,W1,b1,W2,b2).
// ---------------------------------------------------------------------------

#define N_ATOMS   10000
#define N_EDGES   320000
#define N_GRAPHS  32
#define EMB       64
#define LN2F      0.69314718055994530942f
#define PI_F      3.14159265358979323846f

typedef __bf16 bf16;
typedef __attribute__((ext_vector_type(16))) __bf16 v16bf;
typedef __attribute__((ext_vector_type(8)))  __bf16 v8bf;
typedef __attribute__((ext_vector_type(4)))  __bf16 v4bf;
typedef __attribute__((ext_vector_type(8)))  float  v8f;

__device__ __forceinline__ float sspf(float x) {
  // softplus(x) - ln(2), numerically stable
  return fmaxf(x, 0.0f) + __logf(1.0f + __expf(-fabsf(x))) - LN2F;
}

// ----------------- weight repack: f32 [K,N] -> WMMA-fragment bf16 ----------
// Packed index: (((ntile*(K/32) + ktile)*32 + lane)*16 + j)
//   lane<16 : covers K = ktile*32 + 0..7, 16..23   (j = 0..7, 8..15)
//   lane>=16: covers K = ktile*32 + 8..15, 24..31   (koff=8)
// i.e. element j maps to k = ktile*32 + koff + (j&7) + 16*(j>>3), matching the
// B-operand load in the GEMM below (koff = lane<16 ? 0 : 8, n = lane&15).
__global__ void repack_w_kernel(const float* __restrict__ W,
                                bf16* __restrict__ P, int K, int N) {
  int p = blockIdx.x * blockDim.x + threadIdx.x;
  if (p >= K * N) return;
  int j    = p & 15;
  int lane = (p >> 4) & 31;
  int tile = p >> 9;
  int nkt  = K >> 5;
  int kt   = tile % nkt;
  int nt   = tile / nkt;
  int koff = (lane < 16) ? 0 : 8;
  int k = kt * 32 + koff + (j & 7) + ((j >> 3) << 4);
  int n = nt * 16 + (lane & 15);
  P[p] = (bf16)W[(size_t)k * N + n];
}

// ------------------------------ density ------------------------------------
__global__ void density_edge_kernel(const int* __restrict__ ei,
                                    const float* __restrict__ pos,
                                    const int* __restrict__ Z,
                                    const float* __restrict__ coeffs,
                                    float* __restrict__ dens) {
  int t = blockIdx.x * blockDim.x + threadIdx.x;
  if (t >= N_EDGES * 64) return;
  int e = t >> 6, k = t & 63;
  int s = ei[e], d = ei[N_EDGES + e];
  float x = pos[d * 3 + 0] - pos[s * 3 + 0] + 1e-6f;
  float y = pos[d * 3 + 1] - pos[s * 3 + 1] + 1e-6f;
  float z = pos[d * 3 + 2] - pos[s * 3 + 2] + 1e-6f;
  float rn = sqrtf(x * x + y * y + z * z);
  float dr = rn - (6.0f / 63.0f) * (float)k;
  float ex = __expf(-16.0f * dr * dr);
  float c  = 0.5f + 0.5f * __cosf(PI_F * rn * (1.0f / 6.0f));
  float cut = c * c;
  int za = Z[s], zb = Z[d];
  int lo = min(za, zb), hi = max(za, zb);
  float w = ex * cut * coeffs[lo * 8 + hi];
  float L[10] = {1.0f, z, y, x, z * z, y * z, y * y, x * z, x * y, x * x};
  float* base = dens + (size_t)d * 640 + k * 10;
#pragma unroll
  for (int l = 0; l < 10; ++l) atomicAdd(base + l, L[l] * w);
}

__global__ void density_finalize_kernel(const float* __restrict__ dens,
                                        float* __restrict__ xx) {
  int t = blockIdx.x * blockDim.x + threadIdx.x;
  if (t >= N_ATOMS * 64) return;
  int a = t >> 6, k = t & 63;
  const float* p = dens + (size_t)a * 640 + k * 10;
  float v0 = p[0], v1 = p[1], v2 = p[2], v3 = p[3], v4 = p[4];
  float v5 = p[5], v6 = p[6], v7 = p[7], v8 = p[8], v9 = p[9];
  float* q = xx + (size_t)a * 192 + k * 3;
  q[0] = v0 * v0;
  q[1] = v1 * v1 + v2 * v2 + v3 * v3;
  q[2] = v4 * v4 + 2.0f * v5 * v5 + v6 * v6 + 2.0f * v7 * v7 +
         2.0f * v8 * v8 + v9 * v9;
}

__global__ void build_concat_kernel(const float* __restrict__ xx,
                                    const float* __restrict__ emb,
                                    const int* __restrict__ Z,
                                    float* __restrict__ A, int in_dim) {
  int w = in_dim + EMB;
  int t = blockIdx.x * blockDim.x + threadIdx.x;
  if (t >= N_ATOMS * w) return;
  int a = t / w, c = t - a * w;
  A[t] = (c < in_dim) ? xx[(size_t)a * in_dim + c]
                      : emb[Z[a] * EMB + (c - in_dim)];
}

// ----------------------------- WMMA GEMM -----------------------------------
// 128-thread block = 4 waves; each wave owns one 16-col tile (block: 64 cols,
// 16 rows).  Activations are staged per 32-K chunk into LDS with ssp+bf16
// applied once, then each lane reads its A fragment as two b128 LDS loads.
// B is read from the fragment-packed weight pool as one 32B vector load.
// Requires: K % 32 == 0, N % 64 == 0 (true for every GEMM here).
template <int ACT_IN>
__global__ __launch_bounds__(128) void gemm_bf16_kernel(
    const float* __restrict__ X, const bf16* __restrict__ Wp,
    const float* __restrict__ bias, const float* __restrict__ resid,
    float* __restrict__ Y, int M, int K, int N, int act_out) {
  __shared__ __align__(16) bf16 As[16 * 32];
  int tid = threadIdx.x;
  int lane = tid & 31;
  int wv = tid >> 5;
  int half = (lane >= 16);
  int lr = lane & 15;
  int m0 = blockIdx.y * 16;
  int ntile = blockIdx.x * 4 + wv;
  int n0 = ntile * 16;
  int nkt = K >> 5;
  const bf16* wp = Wp + ((size_t)ntile * nkt) * 512 + lane * 16;

  int fr = tid >> 3;              // A-fill: row 0..15
  int fc = (tid & 7) * 4;         // A-fill: col 0..28 step 4
  const float* xrow = X + (size_t)(m0 + fr) * K + fc;
  const bf16* ar = As + lr * 32 + (half ? 8 : 0);

  v8f acc = {0.f, 0.f, 0.f, 0.f, 0.f, 0.f, 0.f, 0.f};
  for (int kt = 0; kt < nkt; ++kt) {
    // cooperative fill: 128 threads x 4 values = 16x32 chunk
    float4 xv = *(const float4*)(xrow + kt * 32);
    if (ACT_IN) {
      xv.x = sspf(xv.x); xv.y = sspf(xv.y);
      xv.z = sspf(xv.z); xv.w = sspf(xv.w);
    }
    v4bf pk;
    pk[0] = (bf16)xv.x; pk[1] = (bf16)xv.y;
    pk[2] = (bf16)xv.z; pk[3] = (bf16)xv.w;
    *(v4bf*)(As + tid * 4) = pk;
    __syncthreads();

    // A fragment: two 16B LDS vector loads
    v8bf lo = *(const v8bf*)(ar);
    v8bf hi = *(const v8bf*)(ar + 16);
    v16bf a;
#pragma unroll
    for (int j = 0; j < 8; ++j) { a[j] = lo[j]; a[j + 8] = hi[j]; }

    // B fragment: contiguous packed 32B
    v16bf b = *(const v16bf*)(wp + (size_t)kt * 512);

    acc = __builtin_amdgcn_wmma_f32_16x16x32_bf16(false, a, false, b,
                                                  (short)0, acc, false, false);
    __syncthreads();
  }
  int col = n0 + lr;
  float bb = bias[col];
#pragma unroll
  for (int i = 0; i < 8; ++i) {
    int row = m0 + i + (half ? 8 : 0);
    float v = acc[i] + bb;
    if (act_out) v = sspf(v);
    if (resid) v += resid[(size_t)row * N + col];
    Y[(size_t)row * N + col] = v;
  }
}

// --------------------- per-edge RBF message + scatter ----------------------
__global__ __launch_bounds__(32) void edge_msg_kernel(
    const int* __restrict__ ei, const float* __restrict__ pos,
    const float* __restrict__ dj, const bf16* __restrict__ WrbfP,
    const float* __restrict__ brbf, float* __restrict__ out, int outd) {
  int lane = threadIdx.x;
  int half = (lane >= 16);
  int lr = lane & 15;
  int e0 = blockIdx.x * 16;
  int e = e0 + lr;
  int s = ei[e], d = ei[N_EDGES + e];
  float x = pos[d * 3 + 0] - pos[s * 3 + 0] + 1e-6f;
  float y = pos[d * 3 + 1] - pos[s * 3 + 1] + 1e-6f;
  float z = pos[d * 3 + 2] - pos[s * 3 + 2] + 1e-6f;
  float rn = sqrtf(x * x + y * y + z * z);
  const float gstep = 6.0f / 31.0f;
  const float gc = -0.5f / (gstep * gstep);
  v16bf a;
  int kb = half ? 8 : 0;
#pragma unroll
  for (int j = 0; j < 8; ++j) {
    float d1 = rn - gstep * (float)(kb + j);
    float d2 = rn - gstep * (float)(kb + j + 16);
    a[j] = (bf16)__expf(gc * d1 * d1);
    a[j + 8] = (bf16)__expf(gc * d2 * d2);
  }
  for (int n0 = 0; n0 < outd; n0 += 16) {
    // K=32 -> one packed fragment per n-tile: contiguous 32B per lane
    v16bf b = *(const v16bf*)(WrbfP + ((size_t)(n0 >> 4)) * 512 + lane * 16);
    v8f acc = {0.f, 0.f, 0.f, 0.f, 0.f, 0.f, 0.f, 0.f};
    acc = __builtin_amdgcn_wmma_f32_16x16x32_bf16(false, a, false, b,
                                                  (short)0, acc, false, false);
    int col = n0 + lr;
    float bb = brbf[col];
#pragma unroll
    for (int i = 0; i < 8; ++i) {
      int ee = e0 + i + (half ? 8 : 0);
      int ss = ei[ee], dd = ei[N_EDGES + ee];
      float v = (acc[i] + bb) * dj[(size_t)ss * outd + col];
      atomicAdd(out + (size_t)dd * outd + col, v);
    }
  }
}

// ------------------------------ tail kernels -------------------------------
__global__ void masked_add_kernel(const float* __restrict__ src,
                                  const int* __restrict__ Z, int elem,
                                  float* __restrict__ yy) {
  int i = blockIdx.x * blockDim.x + threadIdx.x;
  if (i >= N_ATOMS * 64) return;
  if (Z[i >> 6] == elem) yy[i] += src[i];
}

__global__ void init_out_kernel(const float* __restrict__ eshift,
                                float* __restrict__ out) {
  int i = threadIdx.x;
  if (i < N_GRAPHS) out[i] = eshift[i];
}

__global__ void ef_final_kernel(const float* __restrict__ ef2,
                                const float* __restrict__ W,
                                const float* __restrict__ b,
                                const int* __restrict__ batch,
                                float* __restrict__ out) {
  int a = blockIdx.x * blockDim.x + threadIdx.x;
  if (a >= N_ATOMS) return;
  float acc = b[0];
#pragma unroll
  for (int c = 0; c < 64; ++c)
    acc += sspf(ef2[(size_t)a * 64 + c]) * W[c];
  atomicAdd(out + batch[a], acc);
}

// ------------------------------- host side ---------------------------------
extern "C" void kernel_launch(void* const* d_in, const int* in_sizes, int n_in,
                              void* d_out, int out_size, void* d_ws,
                              size_t ws_size, hipStream_t stream) {
  (void)in_sizes; (void)n_in; (void)out_size; (void)ws_size;
  const float* pos    = (const float*)d_in[0];
  const int*   ei     = (const int*)d_in[1];
  const int*   Z      = (const int*)d_in[2];
  const int*   batch  = (const int*)d_in[3];
  const float* eshift = (const float*)d_in[4];
  const float* coeffs = (const float*)d_in[23];
  const float* emb    = (const float*)d_in[84];

  char* ws = (char*)d_ws;
  float* dens = (float*)(ws + 0);           // 10000*640 f32 (25.6 MB)
  float* A    = (float*)(ws + 0);           // alias: 10000*320 f32
  float* dj   = (float*)(ws + 12800000);    // 10000*256 f32
  float* xx   = (float*)(ws + 25600000);    // 10000*256 f32
  float* outb = (float*)(ws + 35840000);    // 10000*256 f32
  float* t1   = (float*)(ws + 46080000);    // 10000*256 f32
  float* t1b  = (float*)(ws + 51200000);    // 10000*128 f32 (inside t1 region)
  float* yy   = (float*)(ws + 56320000);    // 10000*64 f32
  float* ef1  = (float*)(ws + 58880000);    // 10000*64 f32
  float* ef2  = (float*)(ws + 61440000);    // 10000*64 f32
  bf16*  wpool = (bf16*)(ws + 64000000);    // packed bf16 weight pool

  size_t woff = 0;
  auto cv = [&](int idx, int K, int N) -> bf16* {
    bf16* dst = wpool + woff;
    int n = K * N;
    woff += (size_t)n;
    repack_w_kernel<<<(n + 255) / 256, 256, 0, stream>>>(
        (const float*)d_in[idx], dst, K, N);
    return dst;
  };

  const int CONV_IN[5]  = {192, 256, 256, 128, 128};
  const int CONV_OUT[5] = {256, 256, 128, 128, 128};
  bf16 *Watom[5], *Wnbr[5], *Wlout[5], *Wr1[5], *Wr2[5], *Wrbf[5];
  for (int c = 0; c < 5; ++c) {
    int base = 24 + 12 * c;
    int Kc = CONV_IN[c] + EMB, Oc = CONV_OUT[c];
    Watom[c] = cv(base + 0, Kc, Oc);
    Wnbr[c]  = cv(base + 2, Kc, Oc);
    Wlout[c] = cv(base + 4, Oc, Oc);
    Wr1[c]   = cv(base + 6, Oc, Oc);
    Wr2[c]   = cv(base + 8, Oc, Oc);
    Wrbf[c]  = cv(base + 10, 32, Oc);
  }
  const int ANN_D[4] = {128, 128, 128, 64};
  bf16* annW[3][3];
  for (int t = 0; t < 3; ++t)
    for (int l = 0; l < 3; ++l)
      annW[t][l] = cv(5 + 6 * t + 2 * l, ANN_D[l], ANN_D[l + 1]);
  bf16* efW0 = cv(85, 64, 64);
  bf16* efW1 = cv(87, 64, 64);

  auto gemm = [&](const float* X, const bf16* W, const float* bias,
                  const float* resid, float* Y, int M, int K, int N,
                  int actin, int actout) {
    dim3 g(N / 64, M / 16);
    if (actin)
      gemm_bf16_kernel<1><<<g, 128, 0, stream>>>(X, W, bias, resid, Y, M, K, N, actout);
    else
      gemm_bf16_kernel<0><<<g, 128, 0, stream>>>(X, W, bias, resid, Y, M, K, N, actout);
  };

  // ---- density -> xx [10000, 192] ----
  hipMemsetAsync(dens, 0, (size_t)N_ATOMS * 640 * 4, stream);
  density_edge_kernel<<<(N_EDGES * 64 + 255) / 256, 256, 0, stream>>>(
      ei, pos, Z, coeffs, dens);
  density_finalize_kernel<<<(N_ATOMS * 64 + 255) / 256, 256, 0, stream>>>(dens, xx);

  // ---- 5 message-passing layers ----
  for (int c = 0; c < 5; ++c) {
    int ind = CONV_IN[c], outd = CONV_OUT[c], Kc = ind + EMB;
    int base = 24 + 12 * c;
    const float* b_atom = (const float*)d_in[base + 1];
    const float* b_nbr  = (const float*)d_in[base + 3];
    const float* b_lout = (const float*)d_in[base + 5];
    const float* b_r1   = (const float*)d_in[base + 7];
    const float* b_r2   = (const float*)d_in[base + 9];
    const float* b_rbf  = (const float*)d_in[base + 11];

    build_concat_kernel<<<(N_ATOMS * Kc + 255) / 256, 256, 0, stream>>>(
        xx, emb, Z, A, ind);
    // per-atom neighbor transform (factored out of the edge loop)
    gemm(A, Wnbr[c], b_nbr, nullptr, dj, N_ATOMS, Kc, outd, 1, 1);
    // self term out_ii directly into the accumulator
    gemm(A, Watom[c], b_atom, nullptr, outb, N_ATOMS, Kc, outd, 1, 1);
    // edge messages: rbf WMMA + gather dj[src] + scatter-add to out[dst]
    edge_msg_kernel<<<N_EDGES / 16, 32, 0, stream>>>(ei, pos, dj, Wrbf[c],
                                                     b_rbf, outb, outd);
    // 3 residual blocks
    for (int r = 0; r < 3; ++r) {
      gemm(outb, Wr1[c], b_r1, nullptr, t1, N_ATOMS, outd, outd, 1, 0);
      gemm(t1, Wr2[c], b_r2, outb, outb, N_ATOMS, outd, outd, 1, 0);
    }
    gemm(outb, Wlout[c], b_lout, nullptr, xx, N_ATOMS, outd, outd, 1, 0);
  }

  // ---- per-element ANN heads, masked sum into yy ----
  hipMemsetAsync(yy, 0, (size_t)N_ATOMS * 64 * 4, stream);
  const int ELEMS[3] = {1, 6, 7};
  for (int t = 0; t < 3; ++t) {
    int base = 5 + 6 * t;
    gemm(xx, annW[t][0], (const float*)d_in[base + 1], nullptr, t1, N_ATOMS, 128, 128, 1, 0);
    gemm(t1, annW[t][1], (const float*)d_in[base + 3], nullptr, t1b, N_ATOMS, 128, 128, 1, 0);
    gemm(t1b, annW[t][2], (const float*)d_in[base + 5], nullptr, outb, N_ATOMS, 128, 64, 1, 0);
    masked_add_kernel<<<(N_ATOMS * 64 + 255) / 256, 256, 0, stream>>>(
        outb, Z, ELEMS[t], yy);
  }

  // ---- energy head + per-graph reduction ----
  gemm(yy, efW0, (const float*)d_in[86], nullptr, ef1, N_ATOMS, 64, 64, 1, 0);
  gemm(ef1, efW1, (const float*)d_in[88], nullptr, ef2, N_ATOMS, 64, 64, 1, 0);
  init_out_kernel<<<1, 32, 0, stream>>>(eshift, (float*)d_out);
  ef_final_kernel<<<(N_ATOMS + 255) / 256, 256, 0, stream>>>(
      ef2, (const float*)d_in[89], (const float*)d_in[90], batch, (float*)d_out);
}
